// ExtendedQSUP_31653908972057
// MI455X (gfx1250) — matrix operand order repaired and checked by reading, not compiled, not run on hardware
//
#include <hip/hip_runtime.h>
#include <hip/hip_bf16.h>
#include <cmath>

typedef __bf16 bf16;
typedef __attribute__((ext_vector_type(16))) __bf16 v16bf;
typedef __attribute__((ext_vector_type(8)))  __bf16 v8bf;
typedef __attribute__((ext_vector_type(8)))  float  v8f;
typedef __attribute__((ext_vector_type(4)))  unsigned int v4u;
typedef __attribute__((ext_vector_type(8)))  int v8i;
typedef __attribute__((ext_vector_type(4)))  int v4i;

#define BQ 8192
#define INQ 4096
#define HQ 1024
#define PQ 2048
#define NWQ 3
#define NCQ 1000
#define NCPAD 1024

#if defined(__has_builtin)
#  if __has_builtin(__builtin_amdgcn_tensor_load_to_lds)
#    define USE_TDM 1
#  else
#    define USE_TDM 0
#  endif
#else
#  define USE_TDM 0
#endif

// ---------------------------------------------------------------------------
// Block reduction helper (256-thread blocks)
// ---------------------------------------------------------------------------
__device__ inline float block_reduce_sum(float v, float* red) {
    int t = threadIdx.x;
    red[t] = v;
    __syncthreads();
    for (int o = 128; o > 0; o >>= 1) {
        if (t < o) red[t] += red[t + o];
        __syncthreads();
    }
    float r = red[0];
    __syncthreads();
    return r;
}

// ---------------------------------------------------------------------------
// f32 -> bf16 cast (row-major copy, for activations / A operands)
// ---------------------------------------------------------------------------
__global__ void cast_bf16_kernel(const float* __restrict__ in, bf16* __restrict__ out, long n) {
    long i = (long)blockIdx.x * blockDim.x + threadIdx.x;
    if (i < n) out[i] = (bf16)in[i];
}

// ---------------------------------------------------------------------------
// Transpose + cast (+ optional 1/sigma scale) for weight (B) operands.
//   in : K x N f32 (row-major) ; out: Npad x K bf16, rows n>=N zero-filled
// ---------------------------------------------------------------------------
__global__ __launch_bounds__(256) void transpose_cast_kernel(
    const float* __restrict__ in, const float* __restrict__ sigma,
    bf16* __restrict__ out, int K, int N)
{
    __shared__ float tile[32][33];
    const int n0 = blockIdx.x * 32;
    const int k0 = blockIdx.y * 32;
    const int tx = threadIdx.x & 31;
    const int ty = threadIdx.x >> 5;   // 0..7
    float inv = 1.0f;
    if (sigma) inv = 1.0f / sigma[0];

    #pragma unroll
    for (int i = 0; i < 4; ++i) {
        int kk = ty + i * 8;
        int n  = n0 + tx;
        float v = (n < N) ? in[(long)(k0 + kk) * N + n] : 0.0f;
        tile[kk][tx] = v;
    }
    __syncthreads();
    #pragma unroll
    for (int i = 0; i < 4; ++i) {
        int nn = ty + i * 8;
        out[(long)(n0 + nn) * K + (k0 + tx)] = (bf16)(tile[tx][nn] * inv);
    }
}

// ---------------------------------------------------------------------------
// TDM: issue a 2D tile load (tileY rows x 32 bf16) global -> LDS.
// D# per CDNA5 ISA ch.8: group0 {count, lds_addr, global_addr, type=2},
// group1 {data_size=2B, tensor_dim0=32, tensor_dim1=tileY, tile_dim0=32,
//         tile_dim1=tileY, tensor_dim0_stride=ldElems}. Exact-fit tile.
// Wave-level op: EXEC ignored, tracked by TENSORcnt.
// This toolchain's builtin is the 6-arg form:
//   (uint32x4 g0, int32x8 g1, int32x4, int32x4, int32x8, i32 cpol)
// ---------------------------------------------------------------------------
#if USE_TDM
__device__ inline void tdm_load_tile32(unsigned lds_addr, const bf16* gptr,
                                       int tileY, int ldElems)
{
    unsigned long long ga = (unsigned long long)(uintptr_t)gptr;
    v4u g0;
    g0[0] = 1u;                                        // count=1 (valid), user mode
    g0[1] = lds_addr;                                  // LDS byte address
    g0[2] = (unsigned)(ga & 0xFFFFFFFFu);              // global_addr[31:0]
    g0[3] = (unsigned)((ga >> 32) & 0x01FFFFFFu)       // global_addr[56:32]
          | (2u << 30);                                // type = 2 ("image")
    unsigned td0 = 32u;                                // tensor_dim0
    unsigned td1 = (unsigned)tileY;                    // tensor_dim1
    unsigned long long st0 = (unsigned long long)ldElems; // tensor_dim0_stride
    v8i g1;
    g1[0] = (int)(1u << 16);                           // wg_mask=0, data_size=1 (2B)
    g1[1] = (int)((td0 & 0xFFFFu) << 16);              // tensor_dim0[15:0]
    g1[2] = (int)(((td0 >> 16) & 0xFFFFu) | ((td1 & 0xFFFFu) << 16));
    g1[3] = (int)(((td1 >> 16) & 0xFFFFu) | (32u << 16));   // tile_dim0 = 32
    g1[4] = (int)(td1 & 0xFFFFu);                      // tile_dim1 = tileY, tile_dim2=0
    g1[5] = (int)(unsigned)(st0 & 0xFFFFFFFFu);        // dim0_stride[31:0]
    g1[6] = (int)(unsigned)((st0 >> 32) & 0xFFFFu);    // dim0_stride[47:32]
    g1[7] = 0;
    v4i z4; z4[0] = 0; z4[1] = 0; z4[2] = 0; z4[3] = 0;
    v8i z8; z8[0] = 0; z8[1] = 0; z8[2] = 0; z8[3] = 0;
    z8[4] = 0; z8[5] = 0; z8[6] = 0; z8[7] = 0;
    __builtin_amdgcn_tensor_load_to_lds(g0, g1, z4, z4, z8, 0);
}

__device__ inline unsigned lds_byte_addr(const void* p) {
    return (unsigned)(unsigned long long)(uintptr_t)p;
}
#endif

// ---------------------------------------------------------------------------
// Tiled WMMA GEMM: C[M,N] (f32) = A[M,K] (bf16 row-major) @ Bt[N,K]^T (bf16
// row-major, N padded to tile multiple with zero rows).
// 128x128 tile per block, 256 threads = 8 waves (4x2 -> each wave 32x64).
// Double-buffered LDS; tiles fetched by the Tensor Data Mover (wave 0 issues,
// s_wait_tensorcnt gates buffers), overlapping DMA of tile t+1 with WMMA of t.
// Requires: M%128==0, K%64==0 (>=2 steps), lda/ldb multiples of 8.
// ---------------------------------------------------------------------------
__global__ __launch_bounds__(256) void wmma_gemm_bf16_kernel(
    const bf16* __restrict__ A, const bf16* __restrict__ Bt, float* __restrict__ C,
    int M, int N, int K, int lda, int ldb, int ldc,
    long long batchStrideB, long long batchStrideC)
{
    __shared__ __align__(16) bf16 As[2][128 * 32];   // [buf][row][k]
    __shared__ __align__(16) bf16 Bs[2][128 * 32];   // [buf][n][k]

    const int tid  = threadIdx.x;
    const int lane = tid & 31;
    const int wave = tid >> 5;
    const int wm   = wave >> 1;   // 0..3 -> 32-row strip
    const int wn   = wave & 1;    // 0..1 -> 64-col strip
    const int tileM = blockIdx.y * 128;
    const int tileN = blockIdx.x * 128;

    const bf16* Abase = A  + (long long)tileM * lda;
    const bf16* Bbase = Bt + (long long)blockIdx.z * batchStrideB + (long long)tileN * ldb;
    float*      Cb    = C  + (long long)blockIdx.z * batchStrideC;

    v8f acc[2][4];
    #pragma unroll
    for (int mi = 0; mi < 2; ++mi)
        #pragma unroll
        for (int ni = 0; ni < 4; ++ni)
            #pragma unroll
            for (int e = 0; e < 8; ++e) acc[mi][ni][e] = 0.0f;

    const int steps = K >> 5;

#if USE_TDM
    if (wave == 0) {
        tdm_load_tile32(lds_byte_addr(&As[0][0]), Abase, 128, lda);
        tdm_load_tile32(lds_byte_addr(&Bs[0][0]), Bbase, 128, ldb);
    }
#else
    {   // cooperative prologue load of buffer 0
        const int lr  = tid >> 2;
        const int lk8 = (tid & 3) << 3;
        #pragma unroll
        for (int half = 0; half < 2; ++half) {
            int r = lr + half * 64;
            *(v8bf*)&As[0][r * 32 + lk8] = *(const v8bf*)(Abase + (long long)r * lda + lk8);
            *(v8bf*)&Bs[0][r * 32 + lk8] = *(const v8bf*)(Bbase + (long long)r * ldb + lk8);
        }
    }
#endif

    for (int t = 0; t < steps; ++t) {
        const int cur = t & 1;
#if USE_TDM
        if (wave == 0) __builtin_amdgcn_s_wait_tensorcnt(0);
#endif
        __syncthreads();   // buffer `cur` ready; all waves done reading `cur^1`

        if (t + 1 < steps) {
            const int k0n = (t + 1) << 5;
#if USE_TDM
            if (wave == 0) {
                tdm_load_tile32(lds_byte_addr(&As[cur ^ 1][0]), Abase + k0n, 128, lda);
                tdm_load_tile32(lds_byte_addr(&Bs[cur ^ 1][0]), Bbase + k0n, 128, ldb);
            }
#else
            const int lr  = tid >> 2;
            const int lk8 = (tid & 3) << 3;
            #pragma unroll
            for (int half = 0; half < 2; ++half) {
                int r = lr + half * 64;
                *(v8bf*)&As[cur ^ 1][r * 32 + lk8] =
                    *(const v8bf*)(Abase + (long long)r * lda + k0n + lk8);
                *(v8bf*)&Bs[cur ^ 1][r * 32 + lk8] =
                    *(const v8bf*)(Bbase + (long long)r * ldb + k0n + lk8);
            }
#endif
        }

        // ---- fragments per ISA 7.12.2 layouts ----
        v16bf afrag[2], bfrag[4];
        const int ar  = lane & 15;
        const int akb = (lane >> 4) << 3;    // 0 or 8
        #pragma unroll
        for (int mi = 0; mi < 2; ++mi) {
            const bf16* p = &As[cur][(wm * 32 + mi * 16 + ar) * 32];
            v8bf lo = *(const v8bf*)&p[akb];
            v8bf hi = *(const v8bf*)&p[akb + 16];
            afrag[mi] = __builtin_shufflevector(lo, hi,
                0,1,2,3,4,5,6,7,8,9,10,11,12,13,14,15);
        }
        const int bc  = lane & 15;
        const int bkb = (lane >> 4) << 3;    // 0 or 8
        #pragma unroll
        for (int ni = 0; ni < 4; ++ni) {
            const bf16* p = &Bs[cur][(wn * 64 + ni * 16 + bc) * 32 + bkb];
            v8bf lo = *(const v8bf*)&p[0];
            v8bf hi = *(const v8bf*)&p[16];
            bfrag[ni] = __builtin_shufflevector(lo, hi,
                0,1,2,3,4,5,6,7,8,9,10,11,12,13,14,15);
        }
        #pragma unroll
        for (int mi = 0; mi < 2; ++mi)
            #pragma unroll
            for (int ni = 0; ni < 4; ++ni)
                acc[mi][ni] = __builtin_amdgcn_wmma_f32_16x16x32_bf16(
                    false, afrag[mi], false, bfrag[ni],
                    (short)0, acc[mi][ni], false, false);

        __syncthreads();   // all reads of `cur` complete before it is refilled
    }

    // ---- store C: VGPR r -> M = base + (lane<16 ? r : 8+r), N = lane&15 ----
    const int crb = (lane >> 4) << 3;
    const int cc  = lane & 15;
    #pragma unroll
    for (int mi = 0; mi < 2; ++mi) {
        #pragma unroll
        for (int ni = 0; ni < 4; ++ni) {
            int col = tileN + wn * 64 + ni * 16 + cc;
            if (col < N) {
                #pragma unroll
                for (int r = 0; r < 8; ++r) {
                    int row = tileM + wm * 32 + mi * 16 + crb + r;
                    Cb[(long long)row * ldc + col] = acc[mi][ni][r];
                }
            }
        }
    }
}

// ---------------------------------------------------------------------------
// h = gelu(ln(raw + bp)) -> bf16 ; one block per row of length ncols
// ---------------------------------------------------------------------------
__global__ __launch_bounds__(256) void ln_gelu_kernel(
    const float* __restrict__ in, const float* __restrict__ bias,
    const float* __restrict__ g, const float* __restrict__ beta,
    bf16* __restrict__ outbf, int ncols)
{
    __shared__ float red[256];
    const int row = blockIdx.x;
    const float* r = in + (long)row * ncols;
    float s = 0.f, ss = 0.f;
    for (int j = threadIdx.x; j < ncols; j += 256) {
        float v = r[j] + bias[j];
        s += v; ss += v * v;
    }
    float mean = block_reduce_sum(s, red) / ncols;
    float var  = block_reduce_sum(ss, red) / ncols - mean * mean;
    float inv  = rsqrtf(var + 1e-5f);
    for (int j = threadIdx.x; j < ncols; j += 256) {
        float v = r[j] + bias[j];
        float t = (v - mean) * inv * g[j] + beta[j];
        float ge = 0.5f * t * (1.0f + erff(t * 0.70710678118654752f));
        outbf[(long)row * ncols + j] = (bf16)ge;
    }
}

// ---------------------------------------------------------------------------
// Spectral-norm helpers
// ---------------------------------------------------------------------------
__global__ void fill_kernel(float* __restrict__ v, float val, int n) {
    int i = blockIdx.x * blockDim.x + threadIdx.x;
    if (i < n) v[i] = val;
}

__global__ __launch_bounds__(256) void matvec_row_kernel(
    const float* __restrict__ W, const float* __restrict__ u,
    float* __restrict__ y, int n)
{
    __shared__ float red[256];
    const int i = blockIdx.x;
    const float* r = W + (long)i * n;
    float s = 0.f;
    for (int j = threadIdx.x; j < n; j += 256) s += r[j] * u[j];
    float tot = block_reduce_sum(s, red);
    if (threadIdx.x == 0) y[i] = tot;
}

__global__ __launch_bounds__(256) void matvec_col_kernel(
    const float* __restrict__ W, const float* __restrict__ v,
    float* __restrict__ y, int n)
{
    int j = blockIdx.x * 256 + threadIdx.x;
    if (j >= n) return;
    float s = 0.f;
    for (int i = 0; i < n; ++i) s += W[(long)i * n + j] * v[i];
    y[j] = s;
}

__global__ __launch_bounds__(256) void normalize_kernel(float* __restrict__ vec, int n) {
    __shared__ float red[256];
    float s = 0.f;
    for (int i = threadIdx.x; i < n; i += 256) s += vec[i] * vec[i];
    float nrm = sqrtf(block_reduce_sum(s, red)) + 1e-12f;
    for (int i = threadIdx.x; i < n; i += 256) vec[i] /= nrm;
}

__global__ __launch_bounds__(256) void dot_kernel(
    const float* __restrict__ a, const float* __restrict__ b,
    float* __restrict__ out, int n)
{
    __shared__ float red[256];
    float s = 0.f;
    for (int i = threadIdx.x; i < n; i += 256) s += a[i] * b[i];
    float tot = block_reduce_sum(s, red);
    if (threadIdx.x == 0) out[0] = tot;
}

// ---------------------------------------------------------------------------
// Wave stage: per (b,s) row of 2H: +bw, LN(gw,betaw), exp(-x^2), renorm to
// PARTIAL_NORM, phase rotate -> wave_r, wave_i (B,NW,H)
// ---------------------------------------------------------------------------
__global__ __launch_bounds__(256) void wave_stage_kernel(
    const float* __restrict__ gemm, const float* __restrict__ bw,
    const float* __restrict__ gw, const float* __restrict__ betaw,
    const float* __restrict__ phases,
    float* __restrict__ wave_r, float* __restrict__ wave_i)
{
    __shared__ float e[2 * HQ];
    __shared__ float red[256];
    const int b = blockIdx.x, s = blockIdx.y;
    const int n2 = 2 * HQ;
    const float* row = gemm + ((long)b * NWQ + s) * n2;
    const float* bwr = bw + (long)s * n2;
    const float* gwr = gw + (long)s * n2;
    const float* ber = betaw + (long)s * n2;

    float s1 = 0.f, s2 = 0.f;
    for (int j = threadIdx.x; j < n2; j += 256) {
        float v = row[j] + bwr[j];
        s1 += v; s2 += v * v;
    }
    float mean = block_reduce_sum(s1, red) / n2;
    float var  = block_reduce_sum(s2, red) / n2 - mean * mean;
    float inv  = rsqrtf(var + 1e-5f);

    float ps = 0.f;
    for (int j = threadIdx.x; j < n2; j += 256) {
        float v = row[j] + bwr[j];
        float t = (v - mean) * inv * gwr[j] + ber[j];
        float ej = expf(-t * t);
        e[j] = ej;
        ps += ej * ej;
    }
    float nsq = block_reduce_sum(ps, red) + 1e-8f;
    float factor = sqrtf(2.25f / nsq);   // PARTIAL_NORM^2 = 1.5^2
    float ph = phases[s];
    float c = cosf(ph), sn = sinf(ph);
    __syncthreads();
    for (int j = threadIdx.x; j < HQ; j += 256) {
        float a  = e[j] * factor;
        float bt = e[j + HQ] * factor;
        long o = ((long)b * NWQ + s) * HQ + j;
        wave_r[o] = a * c - bt * sn;
        wave_i[o] = a * sn + bt * c;
    }
}

// ---------------------------------------------------------------------------
// Cross-wave combine: cosine similarity softmax over s, weighted sums
// ---------------------------------------------------------------------------
__global__ __launch_bounds__(256) void combine_kernel(
    const float* __restrict__ wave_r, const float* __restrict__ wave_i,
    const float* __restrict__ temperature,
    float* __restrict__ sup_r, float* __restrict__ sup_i)
{
    __shared__ float red[256];
    __shared__ float res[7];
    const int b = blockIdx.x;
    float wr[NWQ][4], wi[NWQ][4];
    float pmsq = 0.f, pn[NWQ] = {0.f,0.f,0.f}, pd[NWQ] = {0.f,0.f,0.f};

    #pragma unroll
    for (int k = 0; k < 4; ++k) {
        int j = threadIdx.x + k * 256;
        #pragma unroll
        for (int s = 0; s < NWQ; ++s) {
            long o = ((long)b * NWQ + s) * HQ + j;
            wr[s][k] = wave_r[o];
            wi[s][k] = wave_i[o];
        }
        float m = (wr[0][k] + wr[1][k] + wr[2][k]) * (1.0f / 3.0f);
        pmsq += m * m;
        #pragma unroll
        for (int s = 0; s < NWQ; ++s) {
            pn[s] += wr[s][k] * wr[s][k];
            pd[s] += wr[s][k] * m;
        }
    }
    float r0 = block_reduce_sum(pmsq, red);
    float r1 = block_reduce_sum(pn[0], red);
    float r2 = block_reduce_sum(pn[1], red);
    float r3 = block_reduce_sum(pn[2], red);
    float r4 = block_reduce_sum(pd[0], red);
    float r5 = block_reduce_sum(pd[1], red);
    float r6 = block_reduce_sum(pd[2], red);
    if (threadIdx.x == 0) {
        res[0]=r0; res[1]=r1; res[2]=r2; res[3]=r3; res[4]=r4; res[5]=r5; res[6]=r6;
    }
    __syncthreads();

    float mn = sqrtf(res[0]) + 1e-8f;
    float invT = 1.0f / temperature[0];
    float l[NWQ];
    float lmax = -1e30f;
    #pragma unroll
    for (int s = 0; s < NWQ; ++s) {
        float wnrm = sqrtf(res[1 + s]) + 1e-8f;
        float cs = res[4 + s] / (wnrm * mn);
        l[s] = cs * invT;
        lmax = fmaxf(lmax, l[s]);
    }
    float esum = 0.f, w[NWQ];
    #pragma unroll
    for (int s = 0; s < NWQ; ++s) { w[s] = expf(l[s] - lmax); esum += w[s]; }
    #pragma unroll
    for (int s = 0; s < NWQ; ++s) w[s] /= esum;

    #pragma unroll
    for (int k = 0; k < 4; ++k) {
        int j = threadIdx.x + k * 256;
        float sr = 0.f, si = 0.f;
        #pragma unroll
        for (int s = 0; s < NWQ; ++s) { sr += wr[s][k] * w[s]; si += wi[s][k] * w[s]; }
        sup_r[(long)b * HQ + j] = sr;
        sup_i[(long)b * HQ + j] = si;
    }
}

// ---------------------------------------------------------------------------
// mag = sqrt(sr^2+si^2+eps) -> bf16
// ---------------------------------------------------------------------------
__global__ void mag_kernel(const float* __restrict__ sr, const float* __restrict__ si,
                           bf16* __restrict__ mag, long n)
{
    long i = (long)blockIdx.x * blockDim.x + threadIdx.x;
    if (i < n) {
        float m = sqrtf(sr[i] * sr[i] + si[i] * si[i] + 1e-8f);
        mag[i] = (bf16)m;
    }
}

// gate = sigmoid(lin+bg); sup *= (gate + 0.1)
__global__ void gate_apply_kernel(const float* __restrict__ lin, const float* __restrict__ bg,
                                  float* __restrict__ sr, float* __restrict__ si,
                                  long n, int H)
{
    long i = (long)blockIdx.x * blockDim.x + threadIdx.x;
    if (i < n) {
        int c = (int)(i % H);
        float g = 1.0f / (1.0f + expf(-(lin[i] + bg[c])));
        sr[i] = sr[i] * g + 0.1f * sr[i];
        si[i] = si[i] * g + 0.1f * si[i];
    }
}

// ---------------------------------------------------------------------------
// top-8 of mag_sq per row -> masked renormalized probs (bf16)
// ---------------------------------------------------------------------------
__global__ __launch_bounds__(256) void topk_probs_kernel(
    const float* __restrict__ sr, const float* __restrict__ si,
    bf16* __restrict__ probs)
{
    __shared__ float msq[HQ];
    __shared__ float work[HQ];
    __shared__ unsigned char sel[HQ];
    __shared__ float rv[256];
    __shared__ int   ri[256];
    const int b = blockIdx.x;
    const int tid = threadIdx.x;

    for (int j = tid; j < HQ; j += 256) {
        float a = sr[(long)b * HQ + j];
        float c = si[(long)b * HQ + j];
        float m = a * a + c * c;
        msq[j] = m; work[j] = m; sel[j] = 0;
    }
    __syncthreads();

    float ksum = 0.f;
    for (int t = 0; t < 8; ++t) {
        float bestv = -1.f; int besti = 1 << 30;
        for (int j = tid; j < HQ; j += 256) {
            float v = work[j];
            if (v > bestv || (v == bestv && j < besti)) { bestv = v; besti = j; }
        }
        rv[tid] = bestv; ri[tid] = besti;
        __syncthreads();
        for (int o = 128; o > 0; o >>= 1) {
            if (tid < o) {
                if (rv[tid + o] > rv[tid] ||
                    (rv[tid + o] == rv[tid] && ri[tid + o] < ri[tid])) {
                    rv[tid] = rv[tid + o]; ri[tid] = ri[tid + o];
                }
            }
            __syncthreads();
        }
        int widx = ri[0]; float wval = rv[0];
        __syncthreads();
        ksum += wval;
        if (tid == 0) { work[widx] = -1.f; sel[widx] = 1; }
        __syncthreads();
    }
    float inv = 1.0f / (ksum + 1e-8f);
    for (int j = tid; j < HQ; j += 256)
        probs[(long)b * HQ + j] = (bf16)(sel[j] ? msq[j] * inv : 0.0f);
}

// gelu(t + bias) -> bf16
__global__ void bias_gelu_bf16_kernel(const float* __restrict__ in, const float* __restrict__ bias,
                                      bf16* __restrict__ out, long n, int ncols)
{
    long i = (long)blockIdx.x * blockDim.x + threadIdx.x;
    if (i < n) {
        float t = in[i] + bias[(int)(i % ncols)];
        out[i] = (bf16)(0.5f * t * (1.0f + erff(t * 0.70710678118654752f)));
    }
}

// out += bias (per column), in place on d_out
__global__ void add_bias_kernel(float* __restrict__ out, const float* __restrict__ bias,
                                long n, int ncols)
{
    long i = (long)blockIdx.x * blockDim.x + threadIdx.x;
    if (i < n) out[i] += bias[(int)(i % ncols)];
}

// ---------------------------------------------------------------------------
// Host driver
// ---------------------------------------------------------------------------
extern "C" void kernel_launch(void* const* d_in, const int* in_sizes, int n_in,
                              void* d_out, int out_size, void* d_ws, size_t ws_size,
                              hipStream_t stream)
{
    const float* x      = (const float*)d_in[0];
    const float* Wp     = (const float*)d_in[1];
    const float* bp     = (const float*)d_in[2];
    const float* gp     = (const float*)d_in[3];
    const float* betap  = (const float*)d_in[4];
    const float* Ww     = (const float*)d_in[5];
    const float* bw     = (const float*)d_in[6];
    const float* gw     = (const float*)d_in[7];
    const float* betaw  = (const float*)d_in[8];
    const float* temp   = (const float*)d_in[9];
    const float* phases = (const float*)d_in[10];
    const float* Wg     = (const float*)d_in[11];
    const float* bg     = (const float*)d_in[12];
    const float* W1     = (const float*)d_in[13];
    const float* b1     = (const float*)d_in[14];
    const float* W2     = (const float*)d_in[15];
    const float* b2     = (const float*)d_in[16];
    (void)in_sizes; (void)n_in; (void)out_size; (void)ws_size;

    char* wsb = (char*)d_ws;
    size_t off = 0;
    auto alloc = [&](size_t bytes) -> void* {
        void* p = wsb + off;
        off += (bytes + 255) & ~(size_t)255;
        return p;
    };

    const long Bn = BQ, INn = INQ, Hn = HQ, Pn = PQ, NWn = NWQ, NCn = NCQ;

    bf16*  x_bf     = (bf16*) alloc(Bn * INn * 2);
    bf16*  WpT      = (bf16*) alloc(Pn * INn * 2);          // P x IN
    float* h_raw    = (float*)alloc(Bn * Pn * 4);
    bf16*  h_bf     = (bf16*) alloc(Bn * Pn * 2);
    bf16*  WwnT     = (bf16*) alloc(NWn * Pn * Pn * 2);     // s: P x P (N x K)
    float* wave_raw = (float*)alloc(Bn * NWn * 2 * Hn * 4);
    float* wave_r   = (float*)alloc(Bn * NWn * Hn * 4);
    float* wave_i   = (float*)alloc(Bn * NWn * Hn * 4);
    float* sup_r    = (float*)alloc(Bn * Hn * 4);
    float* sup_i    = (float*)alloc(Bn * Hn * 4);
    bf16*  mag_bf   = (bf16*) alloc(Bn * Hn * 2);
    float* lin      = (float*)alloc(Bn * Hn * 4);
    bf16*  WgT      = (bf16*) alloc(Hn * Hn * 2);
    bf16*  probs_bf = (bf16*) alloc(Bn * Hn * 2);
    bf16*  W1T      = (bf16*) alloc(Hn * Hn * 2);
    float* t1       = (float*)alloc(Bn * Hn * 4);
    bf16*  h2_bf    = (bf16*) alloc(Bn * Hn * 2);
    bf16*  W2T      = (bf16*) alloc((long)NCPAD * Hn * 2);  // NCpad x H, zero-padded
    float* uvec     = (float*)alloc(Pn * 4);
    float* vvec     = (float*)alloc(Pn * 4);
    float* tvec     = (float*)alloc(Pn * 4);
    float* sigma    = (float*)alloc(4);

    auto blocks = [](long n) { return (unsigned)((n + 255) / 256); };

    // ---- A-operand cast + weight transposes (B operands, Bt[N,K] bf16) ----
    cast_bf16_kernel<<<blocks(Bn * INn), 256, 0, stream>>>(x, x_bf, Bn * INn);
    {   // WpT: (IN x P) -> (P x IN)
        dim3 g((unsigned)(Pn / 32), (unsigned)(INn / 32));
        transpose_cast_kernel<<<g, 256, 0, stream>>>(Wp, nullptr, WpT, (int)INn, (int)Pn);
    }
    {   // WgT, W1T: (H x H) -> (H x H)
        dim3 g((unsigned)(Hn / 32), (unsigned)(Hn / 32));
        transpose_cast_kernel<<<g, 256, 0, stream>>>(Wg, nullptr, WgT, (int)Hn, (int)Hn);
        transpose_cast_kernel<<<g, 256, 0, stream>>>(W1, nullptr, W1T, (int)Hn, (int)Hn);
    }
    {   // W2T: (H x NC) -> (NCpad x H), zero rows NC..NCpad-1
        dim3 g((unsigned)(NCPAD / 32), (unsigned)(Hn / 32));
        transpose_cast_kernel<<<g, 256, 0, stream>>>(W2, nullptr, W2T, (int)Hn, (int)NCn);
    }

    // ---- GEMM1: h_raw = x @ Wp   (8192x2048x4096) ----
    {
        dim3 g((unsigned)(Pn / 128), (unsigned)(Bn / 128), 1);
        wmma_gemm_bf16_kernel<<<g, 256, 0, stream>>>(
            x_bf, WpT, h_raw, (int)Bn, (int)Pn, (int)INn,
            (int)INn, (int)INn, (int)Pn, 0, 0);
    }
    ln_gelu_kernel<<<(unsigned)Bn, 256, 0, stream>>>(h_raw, bp, gp, betap, h_bf, (int)Pn);

    // ---- spectral normalization of Ww (3x 2048x2048, 5 power iterations) ----
    const float u0 = 1.0f / sqrtf((float)Pn);
    for (int s = 0; s < NWQ; ++s) {
        const float* Ws = Ww + (long)s * Pn * Pn;
        fill_kernel<<<blocks(Pn), 256, 0, stream>>>(uvec, u0, (int)Pn);
        for (int it = 0; it < 5; ++it) {
            matvec_row_kernel<<<(unsigned)Pn, 256, 0, stream>>>(Ws, uvec, vvec, (int)Pn);
            normalize_kernel<<<1, 256, 0, stream>>>(vvec, (int)Pn);
            matvec_col_kernel<<<(unsigned)(Pn / 256), 256, 0, stream>>>(Ws, vvec, uvec, (int)Pn);
            normalize_kernel<<<1, 256, 0, stream>>>(uvec, (int)Pn);
        }
        matvec_row_kernel<<<(unsigned)Pn, 256, 0, stream>>>(Ws, uvec, tvec, (int)Pn);
        dot_kernel<<<1, 256, 0, stream>>>(vvec, tvec, sigma, (int)Pn);
        // fused (1/sigma) * transpose -> WwnT[s] (P x P, N x K layout)
        dim3 g((unsigned)(Pn / 32), (unsigned)(Pn / 32));
        transpose_cast_kernel<<<g, 256, 0, stream>>>(
            Ws, sigma, WwnT + (long)s * Pn * Pn, (int)Pn, (int)Pn);
    }

    // ---- GEMM2 (batched over s): wave_raw[b,s,:] = h @ Wwn[s] ----
    {
        dim3 g((unsigned)(Pn / 128), (unsigned)(Bn / 128), (unsigned)NWn);
        wmma_gemm_bf16_kernel<<<g, 256, 0, stream>>>(
            h_bf, WwnT, wave_raw, (int)Bn, (int)Pn, (int)Pn,
            (int)Pn, (int)Pn, (int)(NWn * Pn),
            (long long)(Pn * Pn), (long long)Pn);
    }

    // ---- wave stage + combine ----
    {
        dim3 g((unsigned)Bn, (unsigned)NWn);
        wave_stage_kernel<<<g, 256, 0, stream>>>(wave_raw, bw, gw, betaw, phases, wave_r, wave_i);
    }
    combine_kernel<<<(unsigned)Bn, 256, 0, stream>>>(wave_r, wave_i, temp, sup_r, sup_i);

    // ---- 2 gate steps: mag -> GEMM -> sigmoid gate ----
    for (int step = 0; step < 2; ++step) {
        mag_kernel<<<blocks(Bn * Hn), 256, 0, stream>>>(sup_r, sup_i, mag_bf, Bn * Hn);
        dim3 g((unsigned)(Hn / 128), (unsigned)(Bn / 128), 1);
        wmma_gemm_bf16_kernel<<<g, 256, 0, stream>>>(
            mag_bf, WgT, lin, (int)Bn, (int)Hn, (int)Hn,
            (int)Hn, (int)Hn, (int)Hn, 0, 0);
        gate_apply_kernel<<<blocks(Bn * Hn), 256, 0, stream>>>(
            lin, bg, sup_r, sup_i, Bn * Hn, (int)Hn);
    }

    // ---- top-8 probs ----
    topk_probs_kernel<<<(unsigned)Bn, 256, 0, stream>>>(sup_r, sup_i, probs_bf);

    // ---- head: gelu(probs@W1 + b1) @ W2 + b2 ----
    {
        dim3 g((unsigned)(Hn / 128), (unsigned)(Bn / 128), 1);
        wmma_gemm_bf16_kernel<<<g, 256, 0, stream>>>(
            probs_bf, W1T, t1, (int)Bn, (int)Hn, (int)Hn,
            (int)Hn, (int)Hn, (int)Hn, 0, 0);
    }
    bias_gelu_bf16_kernel<<<blocks(Bn * Hn), 256, 0, stream>>>(t1, b1, h2_bf, Bn * Hn, (int)Hn);
    {
        dim3 g((unsigned)(NCPAD / 128), (unsigned)(Bn / 128), 1);
        wmma_gemm_bf16_kernel<<<g, 256, 0, stream>>>(
            h2_bf, W2T, (float*)d_out, (int)Bn, (int)NCn, (int)Hn,
            (int)Hn, (int)Hn, (int)NCn, 0, 0);
    }
    add_bias_kernel<<<blocks(Bn * NCn), 256, 0, stream>>>((float*)d_out, b2, Bn * NCn, (int)NCn);
}
